// CrossSSM_910533066820
// MI455X (gfx1250) — matrix operand order repaired and checked by reading, not compiled, not run on hardware
//
#include <hip/hip_runtime.h>
#include <hip/hip_bf16.h>

// ---------------------------------------------------------------------------
// Types for WMMA
// ---------------------------------------------------------------------------
typedef __bf16 bf16_t;
typedef __bf16 bf16x16 __attribute__((ext_vector_type(16)));
typedef __bf16 bf16x8  __attribute__((ext_vector_type(8)));
typedef float  floatx8 __attribute__((ext_vector_type(8)));

#define L_SEQ   1024
#define D_MODEL 1024
#define D_INNER 2048
#define E2      4096   // 2*D_INNER
#define D_STATE 16
#define DT_RANK 64
#define NX      96     // DT_RANK + 2*D_STATE

// ---------------------------------------------------------------------------
// fp32 -> bf16 conversion
// ---------------------------------------------------------------------------
__global__ void k_cvt_bf16(const float* __restrict__ src,
                           __hip_bfloat16* __restrict__ dst, int n) {
  int i = blockIdx.x * blockDim.x + threadIdx.x;
  if (i < n) dst[i] = __float2bfloat16(src[i]);
}

// ---------------------------------------------------------------------------
// Generic bf16 WMMA GEMM:  Y[M,N] = X[M,K] * W[N,K]^T   (f32 accumulate)
//   X row stride = ldx, W row stride = ldw. K must be a multiple of 32.
//   M must be a multiple of 64. N handled by clamping B-row index (each
//   output column depends only on its own B column, so clamped loads only
//   pollute columns that are never stored). Optional bf16 secondary output.
//
// Tile mapping: block = 8 waves as 4(M) x 2(N); each wave -> 16(M) x 64(N)
// with 4 accumulators reusing one A fragment -> block tile 64 x 128.
// Lane layouts used here (contiguous k, same for A and B):
//   A (16x32 bf16): lane = (h=lane>>4, m=lane&15); elem e -> K=16h+e, row m
//   B (32x16 bf16): lane = (h, n=lane&15); elem e -> K=16h+e, col n
//   C (16x16 f32) : vgpr v -> row v+8h, col n
// ---------------------------------------------------------------------------
__global__ void k_gemm_bf16(const __hip_bfloat16* __restrict__ X, int ldx,
                            const __hip_bfloat16* __restrict__ W, int ldw,
                            float* __restrict__ Y,
                            __hip_bfloat16* __restrict__ Ybf,
                            int M, int N, int K) {
  const bf16_t* Xp = reinterpret_cast<const bf16_t*>(X);
  const bf16_t* Wp = reinterpret_cast<const bf16_t*>(W);

  const int lane  = threadIdx.x & 31;
  const int wave  = threadIdx.x >> 5;   // 0..7
  const int mwave = wave & 3;           // 4 waves along M
  const int ngrp  = wave >> 2;          // 2 wave-groups along N
  const int m0 = blockIdx.x * 64 + mwave * 16;
  const int n0 = blockIdx.y * 128 + ngrp * 64;

  const int h  = lane >> 4;    // half-wave selector
  const int mr = lane & 15;    // row of A tile / col of B tile

  floatx8 acc[4] = {};

  // Per-tile B base pointers with clamped (always in-bounds) column index.
  int col[4];
  const bf16_t* bp[4];
#pragma unroll
  for (int t = 0; t < 4; ++t) {
    col[t] = n0 + 16 * t + mr;
    const int cc = (col[t] < N) ? col[t] : (N - 1);
    bp[t] = Wp + (size_t)cc * ldw + 16 * h;
  }
  const bf16_t* ap = Xp + (size_t)(m0 + mr) * ldx + 16 * h;

  for (int k0 = 0; k0 < K; k0 += 32) {
    // ---- A fragment: two contiguous 8-elem halves of row m0+mr ----
    bf16x8 a0 = *(const bf16x8*)(ap + k0);
    bf16x8 a1 = *(const bf16x8*)(ap + k0 + 8);
    bf16x16 a;
#pragma unroll
    for (int i = 0; i < 8; ++i) { a[i] = a0[i]; a[8 + i] = a1[i]; }

    __builtin_prefetch(ap + k0 + 128, 0, 1);
    __builtin_prefetch(bp[0] + k0 + 128, 0, 1);

    // ---- 4 B fragments, 4 WMMAs, A reused ----
#pragma unroll
    for (int t = 0; t < 4; ++t) {
      bf16x16 b = *(const bf16x16*)(bp[t] + k0);
      acc[t] = __builtin_amdgcn_wmma_f32_16x16x32_bf16(false, a, false, b,
                                                       (short)0, acc[t],
                                                       false, false);
    }
  }

#pragma unroll
  for (int t = 0; t < 4; ++t) {
    if (col[t] < N) {
#pragma unroll
      for (int v = 0; v < 8; ++v) {
        const int row = m0 + v + 8 * h;
        Y[(size_t)row * N + col[t]] = acc[t][v];
        if (Ybf) Ybf[(size_t)row * N + col[t]] = __float2bfloat16(acc[t][v]);
      }
    }
  }
}

// ---------------------------------------------------------------------------
// Depthwise causal conv1d (k=4) + bias + SiLU.
//   input  xz[l*ld + choff + c], l in [0,L), c in [0,C)
//   output co_f32[l*C + c] and co_bf[l*C + c]
// ---------------------------------------------------------------------------
__global__ void k_conv_silu(const float* __restrict__ xz, int ld, int choff,
                            const float* __restrict__ w,
                            const float* __restrict__ b,
                            float* __restrict__ co,
                            __hip_bfloat16* __restrict__ co_bf,
                            int L, int C) {
  int c = blockIdx.x * blockDim.x + threadIdx.x;
  int l = blockIdx.y;
  if (c >= C) return;
  float acc = b[c];
#pragma unroll
  for (int j = 0; j < 4; ++j) {
    int ll = l - 3 + j;
    if (ll >= 0) acc += w[c * 4 + j] * xz[(size_t)ll * ld + choff + c];
  }
  float s = acc / (1.0f + __expf(-acc));   // silu
  co[(size_t)l * C + c]    = s;
  co_bf[(size_t)l * C + c] = __float2bfloat16(s);
}

// ---------------------------------------------------------------------------
// Selective scan (one branch). lane = (state, sub-channel); 16 lanes per
// channel, 2 channels per wave32, 8 waves per block -> 16 channels/block.
//   delta[l*C + ch], u[l*C + ch], xdbl[l*96 + (64+st | 80+st)]
//   ys[l*C + ch] = sum_n h[n]*C_t[n]
// ---------------------------------------------------------------------------
__global__ void k_scan(const float* __restrict__ delta,
                       const float* __restrict__ u,
                       const float* __restrict__ xdbl,
                       const float* __restrict__ A_log,
                       float* __restrict__ ys,
                       int L, int C) {
  const int lane = threadIdx.x & 31;
  const int wv   = threadIdx.x >> 5;
  const int st   = lane & 15;        // state index 0..15
  const int sub  = lane >> 4;        // which of 2 channels in the wave
  const int ch   = blockIdx.x * 16 + wv * 2 + sub;
  if (ch >= C) return;

  const float A = -__expf(A_log[ch * D_STATE + st]);
  float h = 0.0f;

  for (int l = 0; l < L; ++l) {
    float d = delta[(size_t)l * C + ch];
    d = (d > 20.0f) ? d : log1pf(__expf(d));          // softplus
    const float uu = u[(size_t)l * C + ch];
    const float* row = xdbl + (size_t)l * NX;
    const float bv = row[DT_RANK + st];
    const float cv = row[DT_RANK + D_STATE + st];
    const float dA = __expf(d * A);
    h = dA * h + d * uu * bv;
    float p = h * cv;
    // 16-lane reduction within the channel's lane group
    p += __shfl_xor(p, 1, 16);
    p += __shfl_xor(p, 2, 16);
    p += __shfl_xor(p, 4, 16);
    p += __shfl_xor(p, 8, 16);
    if (st == 0) ys[(size_t)l * C + ch] = p;
  }
}

// ---------------------------------------------------------------------------
// Combine: result = silu(z) * (ys0 + ys1 + D0*u0 + D1*u1)  -> bf16
//   z = xz columns [2048, 4096)
// ---------------------------------------------------------------------------
__global__ void k_combine(const float* __restrict__ xz,
                          const float* __restrict__ ys0,
                          const float* __restrict__ ys1,
                          const float* __restrict__ u0,
                          const float* __restrict__ u1,
                          const float* __restrict__ D0,
                          const float* __restrict__ D1,
                          __hip_bfloat16* __restrict__ res,
                          int L, int C) {
  int c = blockIdx.x * blockDim.x + threadIdx.x;
  int l = blockIdx.y;
  if (c >= C) return;
  const size_t i = (size_t)l * C + c;
  const float zv = xz[(size_t)l * E2 + D_INNER + c];
  const float sz = zv / (1.0f + __expf(-zv));
  const float r  = sz * (ys0[i] + ys1[i] + D0[c] * u0[i] + D1[c] * u1[i]);
  res[i] = __float2bfloat16(r);
}

// ---------------------------------------------------------------------------
// Host-side launch
// ---------------------------------------------------------------------------
static inline size_t align_up(size_t x) { return (x + 255) & ~(size_t)255; }

extern "C" void kernel_launch(void* const* d_in, const int* in_sizes, int n_in,
                              void* d_out, int out_size, void* d_ws, size_t ws_size,
                              hipStream_t stream) {
  (void)in_sizes; (void)n_in; (void)out_size; (void)ws_size;

  const float* x        = (const float*)d_in[0];   // [L, DM]
  const float* x1       = (const float*)d_in[1];
  const float* in_w     = (const float*)d_in[2];   // [4096, 1024]
  const float* in1_w    = (const float*)d_in[3];   // [2048, 1024]
  const float* conv_w   = (const float*)d_in[4];   // [2048, 1, 4]
  const float* conv_b   = (const float*)d_in[5];
  const float* conv1_w  = (const float*)d_in[6];
  const float* conv1_b  = (const float*)d_in[7];
  const float* xp_w     = (const float*)d_in[8];   // [96, 2048]
  const float* xp1_w    = (const float*)d_in[9];
  const float* dt_w     = (const float*)d_in[10];  // [2048, 64]
  const float* dt1_w    = (const float*)d_in[11];
  const float* A_log    = (const float*)d_in[12];  // [2048, 16]
  const float* A_log1   = (const float*)d_in[13];
  const float* Dsk      = (const float*)d_in[14];
  const float* Dsk1     = (const float*)d_in[15];
  const float* out_w    = (const float*)d_in[16];  // [1024, 2048]
  float* out            = (float*)d_out;           // [L, DM] fp32

  char* ws = (char*)d_ws;
  size_t off = 0;
  auto alloc = [&](size_t bytes) -> void* {
    void* p = ws + off; off = align_up(off + bytes); return p;
  };

  typedef __hip_bfloat16 hbf;
  hbf* x_bf    = (hbf*)alloc((size_t)L_SEQ * D_MODEL * 2);
  hbf* x1_bf   = (hbf*)alloc((size_t)L_SEQ * D_MODEL * 2);
  hbf* win_bf  = (hbf*)alloc((size_t)E2 * D_MODEL * 2);
  hbf* win1_bf = (hbf*)alloc((size_t)D_INNER * D_MODEL * 2);
  hbf* wxp_bf  = (hbf*)alloc((size_t)NX * D_INNER * 2);
  hbf* wxp1_bf = (hbf*)alloc((size_t)NX * D_INNER * 2);
  hbf* wdt_bf  = (hbf*)alloc((size_t)D_INNER * DT_RANK * 2);
  hbf* wdt1_bf = (hbf*)alloc((size_t)D_INNER * DT_RANK * 2);
  hbf* wout_bf = (hbf*)alloc((size_t)D_MODEL * D_INNER * 2);

  float* xz     = (float*)alloc((size_t)L_SEQ * E2 * 4);
  float* x1p    = (float*)alloc((size_t)L_SEQ * D_INNER * 4);
  float* co     = (float*)alloc((size_t)L_SEQ * D_INNER * 4);
  hbf*   co_bf  = (hbf*)  alloc((size_t)L_SEQ * D_INNER * 2);
  float* co1    = (float*)alloc((size_t)L_SEQ * D_INNER * 4);
  hbf*   co1_bf = (hbf*)  alloc((size_t)L_SEQ * D_INNER * 2);
  float* xd     = (float*)alloc((size_t)L_SEQ * NX * 4);
  hbf*   xd_bf  = (hbf*)  alloc((size_t)L_SEQ * NX * 2);
  float* xd1    = (float*)alloc((size_t)L_SEQ * NX * 4);
  hbf*   xd1_bf = (hbf*)  alloc((size_t)L_SEQ * NX * 2);
  float* dlt    = (float*)alloc((size_t)L_SEQ * D_INNER * 4);
  float* dlt1   = (float*)alloc((size_t)L_SEQ * D_INNER * 4);
  float* ys0    = (float*)alloc((size_t)L_SEQ * D_INNER * 4);
  float* ys1    = (float*)alloc((size_t)L_SEQ * D_INNER * 4);
  hbf*   res_bf = (hbf*)  alloc((size_t)L_SEQ * D_INNER * 2);

  auto cvt = [&](const float* s, hbf* d, int n) {
    k_cvt_bf16<<<(n + 255) / 256, 256, 0, stream>>>(s, d, n);
  };
  auto gemm = [&](const hbf* X, int ldx, const hbf* W, int ldw,
                  float* Y, hbf* Ybf, int M, int N, int K) {
    dim3 g(M / 64, (N + 127) / 128);
    k_gemm_bf16<<<g, 256, 0, stream>>>(X, ldx, W, ldw, Y, Ybf, M, N, K);
  };

  // 1) conversions
  cvt(x,     x_bf,    L_SEQ * D_MODEL);
  cvt(x1,    x1_bf,   L_SEQ * D_MODEL);
  cvt(in_w,  win_bf,  E2 * D_MODEL);
  cvt(in1_w, win1_bf, D_INNER * D_MODEL);
  cvt(xp_w,  wxp_bf,  NX * D_INNER);
  cvt(xp1_w, wxp1_bf, NX * D_INNER);
  cvt(dt_w,  wdt_bf,  D_INNER * DT_RANK);
  cvt(dt1_w, wdt1_bf, D_INNER * DT_RANK);
  cvt(out_w, wout_bf, D_MODEL * D_INNER);

  // 2) input projections (WMMA GEMMs)
  gemm(x_bf,  D_MODEL, win_bf,  D_MODEL, xz,  nullptr, L_SEQ, E2,      D_MODEL);
  gemm(x1_bf, D_MODEL, win1_bf, D_MODEL, x1p, nullptr, L_SEQ, D_INNER, D_MODEL);

  // 3) depthwise causal conv + SiLU
  {
    dim3 g((D_INNER + 255) / 256, L_SEQ);
    k_conv_silu<<<g, 256, 0, stream>>>(xz,  E2,      0, conv_w,  conv_b,
                                       co,  co_bf,  L_SEQ, D_INNER);
    k_conv_silu<<<g, 256, 0, stream>>>(x1p, D_INNER, 0, conv1_w, conv1_b,
                                       co1, co1_bf, L_SEQ, D_INNER);
  }

  // 4) x_dbl = co . x_proj_w^T   (N = 96)
  gemm(co_bf,  D_INNER, wxp_bf,  D_INNER, xd,  xd_bf,  L_SEQ, NX, D_INNER);
  gemm(co1_bf, D_INNER, wxp1_bf, D_INNER, xd1, xd1_bf, L_SEQ, NX, D_INNER);

  // 5) delta = x_dbl[:, :64] . dt_proj_w^T   (K = 64, ldx = 96)
  gemm(xd_bf,  NX, wdt_bf,  DT_RANK, dlt,  nullptr, L_SEQ, D_INNER, DT_RANK);
  gemm(xd1_bf, NX, wdt1_bf, DT_RANK, dlt1, nullptr, L_SEQ, D_INNER, DT_RANK);

  // 6) selective scans (per-channel recurrence, lane = state)
  {
    dim3 g(D_INNER / 16);
    k_scan<<<g, 256, 0, stream>>>(dlt,  co,  xd,  A_log,  ys0, L_SEQ, D_INNER);
    k_scan<<<g, 256, 0, stream>>>(dlt1, co1, xd1, A_log1, ys1, L_SEQ, D_INNER);
  }

  // 7) gate + skip + branch sum
  {
    dim3 g((D_INNER + 255) / 256, L_SEQ);
    k_combine<<<g, 256, 0, stream>>>(xz, ys0, ys1, co, co1, Dsk, Dsk1,
                                     res_bf, L_SEQ, D_INNER);
  }

  // 8) output projection -> d_out (fp32)
  gemm(res_bf, D_INNER, wout_bf, D_INNER, out, nullptr, L_SEQ, D_MODEL, D_INNER);
}
